// DotProductAttention_70669391888880
// MI455X (gfx1250) — matrix operand order repaired and checked by reading, not compiled, not run on hardware
//
#include <hip/hip_runtime.h>
#include <stdint.h>

// ---------------------------------------------------------------------------
// DotProductAttention for MI455X (gfx1250): FA2-style flash attention.
//   Q,K,V: [32, 2048, 64] f32  ->  O: [32, 2048, 64] f32
// Pass 1: convert Q/K/V to f16 in workspace. Q is pre-scaled by
//         (1/sqrt(d)) * log2(e) so softmax runs in exp2 domain (no per-score
//         multiply before v_exp_f32).
// Pass 2: flash attention. Each wave owns 32 query rows (two 16-row M-tiles)
//         so every K/V B-fragment loaded from LDS feeds two
//         v_wmma_f32_16x16x32_f16 ops (32 WMMAs per KV-64 tile). f32 online
//         softmax in A-layout (one xor-16 exchange per reduction), packed
//         f32->f16 conversion via v_cvt_pk_rtz_f16_f32.
//         Double-buffered async K staging overlapped with compute.
// ---------------------------------------------------------------------------

typedef _Float16 h4   __attribute__((ext_vector_type(4)));
typedef __fp16   f16x2 __attribute__((ext_vector_type(2)));   // cvt_pkrtz result type
typedef _Float16 v8h  __attribute__((ext_vector_type(8)));
typedef _Float16 v16h __attribute__((ext_vector_type(16)));
typedef float    v8f  __attribute__((ext_vector_type(8)));

#define N_HEADS 32
#define SEQ     2048
#define DIM     64
#define KVT     64            // kv tile width
#define NITER   (SEQ / KVT)   // 32

// ---------------- pass 1: f32 -> f16 conversion (optional scale) -----------
__global__ void cvt_f32_to_f16_kernel(const float* __restrict__ src,
                                      _Float16* __restrict__ dst,
                                      float scale, int n4) {
  int i = blockIdx.x * blockDim.x + threadIdx.x;
  if (i < n4) {
    float4 f = ((const float4*)src)[i];
    h4 h = { (_Float16)(f.x * scale), (_Float16)(f.y * scale),
             (_Float16)(f.z * scale), (_Float16)(f.w * scale) };
    ((h4*)dst)[i] = h;
  }
}

// ---------------- pass 2: flash attention ----------------------------------
// Block: 128 threads = 4 waves; each wave: 32 q rows; block: 128 q rows.
__launch_bounds__(128)
__global__ void flash_attn_kernel(const _Float16* __restrict__ Qh,
                                  const _Float16* __restrict__ Kh,
                                  const _Float16* __restrict__ Vh,
                                  float* __restrict__ out) {
  __shared__ __align__(32) _Float16 sK [2][KVT * 64];   // [buf][kv][d]
  __shared__ __align__(32) _Float16 sVt[2][64 * KVT];   // [buf][d][kv]
  __shared__ __align__(16) float    sS [4][32 * 68];    // per-wave 32x64 scores, padded

  const int tid  = threadIdx.x;
  const int lane = tid & 31;
  const int wave = tid >> 5;
  const int grp  = lane >> 4;   // lane group: 0 = lanes 0-15, 1 = lanes 16-31
  const int ln   = lane & 15;

  const int head  = blockIdx.y;
  const int qrow0 = blockIdx.x * 128 + wave * 32;
  const size_t headBase = (size_t)head * SEQ * DIM;

  // staging decomposition: 128 threads cover a 64x64 f16 tile, 32 halfs each
  const int sr = tid >> 1;          // kv row 0..63
  const int sc = (tid & 1) * 32;    // d col 0 or 32

  // --- Q fragments: A-matrix layout (16x32 f16); [mt][dchunk], d=0..63.
  v16h qa[2][2];
  #pragma unroll
  for (int mt = 0; mt < 2; ++mt) {
    const _Float16* qrow = Qh + headBase + (size_t)(qrow0 + mt * 16 + ln) * DIM;
    #pragma unroll
    for (int f = 0; f < 2; ++f) {
      const _Float16* p = qrow + f * 32 + grp * 16;
      v8h lo = *(const v8h*)(p);
      v8h hi = *(const v8h*)(p + 8);
      qa[mt][f] = __builtin_shufflevector(lo, hi, 0,1,2,3,4,5,6,7,8,9,10,11,12,13,14,15);
    }
  }

  v8f acc[2][4] = {};
  float m_i[2] = { -3.0e38f, -3.0e38f };   // softmax state (log2 domain), row = ln
  float l_i[2] = { 0.0f, 0.0f };

  // ---- preload tile 0: async K -> LDS, V -> regs -> transposed LDS --------
  {
    const size_t g = headBase + (size_t)sr * DIM + sc;
    const unsigned ldsK = (unsigned)(uintptr_t)(&sK[0][sr * 64 + sc]);
    const uint64_t gK   = (uint64_t)(uintptr_t)(Kh + g);
    asm volatile("global_load_async_to_lds_b128 %0, %1, off"           :: "v"(ldsK), "v"(gK) : "memory");
    asm volatile("global_load_async_to_lds_b128 %0, %1, off offset:16" :: "v"(ldsK), "v"(gK) : "memory");
    asm volatile("global_load_async_to_lds_b128 %0, %1, off offset:32" :: "v"(ldsK), "v"(gK) : "memory");
    asm volatile("global_load_async_to_lds_b128 %0, %1, off offset:48" :: "v"(ldsK), "v"(gK) : "memory");
    #pragma unroll
    for (int q = 0; q < 4; ++q) {
      v8h v = *(const v8h*)(Vh + g + q * 8);
      #pragma unroll
      for (int j = 0; j < 8; ++j)
        sVt[0][(sc + q * 8 + j) * KVT + sr] = v[j];
    }
  }

  for (int it = 0; it < NITER; ++it) {
    const int buf = it & 1;
    const bool hasNext = (it + 1 < NITER);

    asm volatile("s_wait_asynccnt 0" ::: "memory");  // my K(tile it) in LDS
    asm volatile("s_wait_dscnt 0"    ::: "memory");  // my V(tile it) in LDS
    __syncthreads();                                 // everyone's tile it ready,
                                                     // everyone done with buf^1

    // ---- issue stage for tile it+1 into buf^1 (overlaps with compute) -----
    v8h nv[4] = {};
    if (hasNext) {
      const size_t g = headBase + (size_t)((it + 1) * KVT + sr) * DIM + sc;
      const unsigned ldsK = (unsigned)(uintptr_t)(&sK[buf ^ 1][sr * 64 + sc]);
      const uint64_t gK   = (uint64_t)(uintptr_t)(Kh + g);
      asm volatile("global_load_async_to_lds_b128 %0, %1, off"           :: "v"(ldsK), "v"(gK) : "memory");
      asm volatile("global_load_async_to_lds_b128 %0, %1, off offset:16" :: "v"(ldsK), "v"(gK) : "memory");
      asm volatile("global_load_async_to_lds_b128 %0, %1, off offset:32" :: "v"(ldsK), "v"(gK) : "memory");
      asm volatile("global_load_async_to_lds_b128 %0, %1, off offset:48" :: "v"(ldsK), "v"(gK) : "memory");
      #pragma unroll
      for (int q = 0; q < 4; ++q) nv[q] = *(const v8h*)(Vh + g + q * 8);
      if (it + 2 < NITER) {                          // warm L2 for tile it+2
        __builtin_prefetch(Kh + g + KVT * DIM, 0, 1);
        __builtin_prefetch(Vh + g + KVT * DIM, 0, 1);
      }
    }

    // ---- S = Q' K^T (log2-domain logits): 2 M-tiles x 4 N-chunks ----------
    // Each K B-fragment pair feeds both M-tiles.
    const _Float16* kb = &sK[buf][0];
    v8f c[2][4];
    #pragma unroll
    for (int nc = 0; nc < 4; ++nc) {
      const _Float16* bp = kb + (nc * 16 + ln) * 64 + grp * 16;
      v16h b0 = *(const v16h*)(bp);         // d = 0..31 contraction half
      v16h b1 = *(const v16h*)(bp + 32);    // d = 32..63 contraction half
      #pragma unroll
      for (int mt = 0; mt < 2; ++mt) {
        v8f z = {};
        z = __builtin_amdgcn_wmma_f32_16x16x32_f16(false, qa[mt][0], false, b0, (short)0, z, false, false);
        c[mt][nc] = __builtin_amdgcn_wmma_f32_16x16x32_f16(false, qa[mt][1], false, b1, (short)0, z, false, false);
      }
    }

    // ---- transpose scores C->A layout through per-wave LDS tile -----------
    float* sw = &sS[wave][0];
    #pragma unroll
    for (int mt = 0; mt < 2; ++mt) {
      #pragma unroll
      for (int r = 0; r < 8; ++r) {
        const int row = mt * 16 + r + grp * 8;   // C layout: M = r + 8*grp
        sw[row * 68 + ln]      = c[mt][0][r];
        sw[row * 68 + 16 + ln] = c[mt][1][r];
        sw[row * 68 + 32 + ln] = c[mt][2][r];
        sw[row * 68 + 48 + ln] = c[mt][3][r];
      }
    }
    asm volatile("s_wait_dscnt 0" ::: "memory");   // own-wave LDS RAW fence

    // ---- online softmax per M-tile (exp2 domain, A layout) ----------------
    v16h pa0[2], pa1[2];                           // P fragments, A layout
    #pragma unroll
    for (int mt = 0; mt < 2; ++mt) {
      float s[32];
      {
        const float* srd = sw + (mt * 16 + ln) * 68 + grp * 16;
        #pragma unroll
        for (int e = 0; e < 8; ++e) {
          s[e]      = srd[e];
          s[8 + e]  = srd[8 + e];
          s[16 + e] = srd[32 + e];
          s[24 + e] = srd[40 + e];
        }
      }
      float tm = s[0];
      #pragma unroll
      for (int e = 1; e < 32; ++e) tm = fmaxf(tm, s[e]);
      tm = fmaxf(tm, __shfl_xor(tm, 16, 32));      // other half of the row

      const float mn   = fmaxf(m_i[mt], tm);
      const float corr = __builtin_amdgcn_exp2f(m_i[mt] - mn);
      float sum = 0.0f;
      union { v16h v; f16x2 h[8]; } u0, u1;
      #pragma unroll
      for (int e = 0; e < 8; ++e) {
        float pA = __builtin_amdgcn_exp2f(s[2 * e]     - mn);
        float pB = __builtin_amdgcn_exp2f(s[2 * e + 1] - mn);
        sum += pA + pB;
        u0.h[e] = __builtin_amdgcn_cvt_pkrtz(pA, pB);  // packed f16 pair
      }
      #pragma unroll
      for (int e = 0; e < 8; ++e) {
        float pA = __builtin_amdgcn_exp2f(s[16 + 2 * e]     - mn);
        float pB = __builtin_amdgcn_exp2f(s[16 + 2 * e + 1] - mn);
        sum += pA + pB;
        u1.h[e] = __builtin_amdgcn_cvt_pkrtz(pA, pB);
      }
      pa0[mt] = u0.v;
      pa1[mt] = u1.v;
      sum += __shfl_xor(sum, 16, 32);
      l_i[mt] = l_i[mt] * corr + sum;
      m_i[mt] = mn;

      // broadcast corr to C-layout accumulator rows (row = r + 8*grp)
      #pragma unroll
      for (int r = 0; r < 8; ++r) {
        const float cc = __shfl(corr, r + 8 * grp, 32);
        #pragma unroll
        for (int d = 0; d < 4; ++d) acc[mt][d][r] *= cc;
      }
    }

    // ---- O += P V : each V B-fragment pair feeds both M-tiles -------------
    const _Float16* vb = &sVt[buf][0];
    #pragma unroll
    for (int d = 0; d < 4; ++d) {
      const _Float16* vp = vb + (d * 16 + ln) * KVT + grp * 16;
      v16h bv0 = *(const v16h*)(vp);        // kv = 0..31 contraction half
      v16h bv1 = *(const v16h*)(vp + 32);   // kv = 32..63 contraction half
      #pragma unroll
      for (int mt = 0; mt < 2; ++mt) {
        acc[mt][d] = __builtin_amdgcn_wmma_f32_16x16x32_f16(false, pa0[mt], false, bv0, (short)0, acc[mt][d], false, false);
        acc[mt][d] = __builtin_amdgcn_wmma_f32_16x16x32_f16(false, pa1[mt], false, bv1, (short)0, acc[mt][d], false, false);
      }
    }

    // ---- commit V(tile it+1) transpose after compute (hides load latency) -
    if (hasNext) {
      _Float16* vt = &sVt[buf ^ 1][0];
      #pragma unroll
      for (int q = 0; q < 4; ++q) {
        #pragma unroll
        for (int j = 0; j < 8; ++j)
          vt[(sc + q * 8 + j) * KVT + sr] = nv[q][j];
      }
    }
  }

  // ---- epilogue: normalize by row sums, write f32 output -------------------
  #pragma unroll
  for (int mt = 0; mt < 2; ++mt) {
    #pragma unroll
    for (int r = 0; r < 8; ++r) {
      const float lr  = __shfl(l_i[mt], r + 8 * grp, 32);
      const float inv = 1.0f / lr;
      const int row = qrow0 + mt * 16 + r + grp * 8;
      float* orow = out + headBase + (size_t)row * DIM + ln;
      #pragma unroll
      for (int d = 0; d < 4; ++d)
        orow[d * 16] = acc[mt][d][r] * inv;
    }
  }
}

// ---------------------------------------------------------------------------
extern "C" void kernel_launch(void* const* d_in, const int* in_sizes, int n_in,
                              void* d_out, int out_size, void* d_ws, size_t ws_size,
                              hipStream_t stream) {
  const float* Q = (const float*)d_in[0];
  const float* K = (const float*)d_in[1];
  const float* V = (const float*)d_in[2];
  float* out = (float*)d_out;

  const size_t N = (size_t)N_HEADS * SEQ * DIM;   // 4,194,304 elements each
  _Float16* Qh = (_Float16*)d_ws;                 // needs 3*N*2 = 24 MiB of ws
  _Float16* Kh = Qh + N;
  _Float16* Vh = Kh + N;

  const int n4 = (int)(N / 4);
  const int cvtBlocks = (n4 + 255) / 256;
  // Q scale = (1/sqrt(64)) * log2(e): softmax runs in exp2 domain.
  cvt_f32_to_f16_kernel<<<cvtBlocks, 256, 0, stream>>>(Q, Qh, 0.125f * 1.44269504088896340736f, n4);
  cvt_f32_to_f16_kernel<<<cvtBlocks, 256, 0, stream>>>(K, Kh, 1.0f, n4);
  cvt_f32_to_f16_kernel<<<cvtBlocks, 256, 0, stream>>>(V, Vh, 1.0f, n4);

  dim3 grid(SEQ / 128, N_HEADS);                  // 16 q-tiles x 32 heads
  flash_attn_kernel<<<grid, 128, 0, stream>>>(Qh, Kh, Vh, out);
}
